// RegionalHead_45561013076490
// MI455X (gfx1250) — matrix operand order repaired and checked by reading, not compiled, not run on hardware
//
#include <hip/hip_runtime.h>
#include <stdint.h>

typedef __bf16 bf16;
typedef __attribute__((ext_vector_type(16))) __bf16 v16bf;
typedef __attribute__((ext_vector_type(8)))  __bf16 v8bf;
typedef __attribute__((ext_vector_type(4)))  __bf16 v4bf;
typedef __attribute__((ext_vector_type(8)))  float  v8f;

#define BDIM 128
#define SDIM 8
#define NGRP (BDIM*SDIM)   // 1024 independent attention problems
#define CDIM 120
#define CP   128
#define LDIM 469
#define LP   480           // 15 k-steps of 32
#define HDIM 64

// ---------------------------------------------------------------------------
// A operand (16x32 bf16, row-major source): lanes 0-15 -> M=lane, K=+0..7 in
// v0..3 and K=+16..23 in v4..7; lanes 16-31 -> M=lane-16, K=+8..15 / +24..31.
__device__ __forceinline__ v16bf load_a16(const bf16* base, int stride,
                                          int row0, int kbase, int lane) {
  int r  = row0 + (lane & 15);
  int k0 = kbase + (((lane >> 4) & 1) << 3);
  const bf16* p = base + r * stride + k0;
  v8bf lo = *(const v8bf*)p;          // K +0..7  (or +8..15)
  v8bf hi = *(const v8bf*)(p + 16);   // K +16..23 (or +24..31)
  return __builtin_shufflevector(lo, hi, 0,1,2,3,4,5,6,7,8,9,10,11,12,13,14,15);
}

// B operand (32x16 bf16): lane holds column N=n0+lane%16; K=+0..7 / +16..23
// (lanes<16) or +8..15 / +24..31 (lanes>=16), matching the A layout,
// from a column-major [N][K] image.
__device__ __forceinline__ v16bf load_b16(const bf16* base, int stride,
                                          int n0, int kbase, int lane) {
  int nn = n0 + (lane & 15);
  int k0 = kbase + (((lane >> 4) & 1) << 3);
  const bf16* p = base + nn * stride + k0;
  v8bf lo = *(const v8bf*)p;
  v8bf hi = *(const v8bf*)(p + 16);
  return __builtin_shufflevector(lo, hi, 0,1,2,3,4,5,6,7,8,9,10,11,12,13,14,15);
}

#define WMMA_BF16(A, B, C) \
  __builtin_amdgcn_wmma_f32_16x16x32_bf16(false, (A), false, (B), (short)0, (C), false, false)

// ---------------------------------------------------------------------------
// Pre-pass: W[L=469][H=64] fp32 (row-major) -> bf16 column-major [64][480],
// zero-padded in K, for all three weights. Lives in d_ws; hot in L2.
__global__ __launch_bounds__(256) void wt_transpose_kernel(
    const float* __restrict__ Wk, const float* __restrict__ Wq,
    const float* __restrict__ Wv, bf16* __restrict__ wt) {
  int i = blockIdx.x * 256 + threadIdx.x;        // over 3*64*480
  if (i >= 3 * HDIM * LP) return;
  int w = i / (HDIM * LP);
  int r = i - w * (HDIM * LP);
  int nIdx = r / LP;
  int k = r - nIdx * LP;
  const float* W = (w == 0) ? Wk : ((w == 1) ? Wq : Wv);
  float v = (k < LDIM) ? W[k * HDIM + nIdx] : 0.0f;
  wt[i] = (bf16)v;
}

// ---------------------------------------------------------------------------
// One workgroup per n: full fused QKV projection + attention + output.
// 8 wave32 = 2 waves/SIMD; ~205KB LDS => 1 block/WGP, so allow half the
// register file per wave (waves_per_eu(2)) to keep operand groups resident.
__global__ __launch_bounds__(256) __attribute__((amdgpu_waves_per_eu(2)))
void attn_kernel(
    const float* __restrict__ x, const bf16* __restrict__ wt,
    float* __restrict__ out) {
  extern __shared__ char smem[];
  bf16* Xb = (bf16*)smem;            // [CP][LP]     61440 el
  bf16* Qb = Xb + CP * LP;           // [CP][HDIM]    8192 el
  bf16* Kb = Qb + CP * HDIM;         // [CP][HDIM]    8192 el
  bf16* VT = Kb + CP * HDIM;         // [HDIM][CP]    8192 el (V transposed)
  bf16* Pb = VT + HDIM * CP;         // [CP][CP]     16384 el
  float* Ob = (float*)smem;          // alias of Xb region after projections

  const int tid  = threadIdx.x;
  const int lane = tid & 31;
  const int wv   = tid >> 5;          // 0..7, M-tile owner (wave32)
  const int n    = blockIdx.x;
  const int half = (lane >> 4) & 1;   // row-half of C/D layout
  const int l15  = lane & 15;

  // ---- stage X_n -> bf16 LDS, zero-padded to [128][480] -------------------
  const float* xs = x + (size_t)n * (CDIM * LDIM);
  for (int j = tid; j < CP * (LP / 4); j += 256) {
    int row = j / (LP / 4);
    int cb  = (j - row * (LP / 4)) * 4;
    float f0 = 0.f, f1 = 0.f, f2 = 0.f, f3 = 0.f;
    if (row < CDIM) {
      const float* rp = xs + row * LDIM;
      if (cb + 3 < LDIM) { f0 = rp[cb]; f1 = rp[cb+1]; f2 = rp[cb+2]; f3 = rp[cb+3]; }
      else {
        if (cb     < LDIM) f0 = rp[cb];
        if (cb + 1 < LDIM) f1 = rp[cb+1];
        if (cb + 2 < LDIM) f2 = rp[cb+2];
      }
    }
    v4bf pk = { (bf16)f0, (bf16)f1, (bf16)f2, (bf16)f3 };
    *(v4bf*)(Xb + row * LP + cb) = pk;
  }
  __syncthreads();

  // ---- Phase 1: Q/K/V = X @ W  (15 k-steps, 4 n-tiles, 3 weights) ---------
  const bf16* WTk = wt;
  const bf16* WTq = wt + HDIM * LP;
  const bf16* WTv = wt + 2 * HDIM * LP;
  const v8f vz = {0.f,0.f,0.f,0.f,0.f,0.f,0.f,0.f};
  v8f accQ[4], accK[4], accV[4];
#pragma unroll
  for (int t = 0; t < 4; ++t) { accQ[t] = vz; accK[t] = vz; accV[t] = vz; }

#pragma unroll 1
  for (int ks = 0; ks < LP / 32; ++ks) {
    v16bf a = load_a16(Xb, LP, wv * 16, ks * 32, lane);
    // Batch all 12 independent B-tile loads; the empty asm is a single
    // ordering point so one wait covers the whole 26-load clause and the
    // 12 WMMAs then issue back-to-back.
    v16bf b[12];
#pragma unroll
    for (int nt = 0; nt < 4; ++nt) {
      b[nt]     = load_b16(WTq, LP, nt * 16, ks * 32, lane);
      b[4 + nt] = load_b16(WTk, LP, nt * 16, ks * 32, lane);
      b[8 + nt] = load_b16(WTv, LP, nt * 16, ks * 32, lane);
    }
    asm volatile("" : "+v"(a),
                      "+v"(b[0]), "+v"(b[1]), "+v"(b[2]),  "+v"(b[3]),
                      "+v"(b[4]), "+v"(b[5]), "+v"(b[6]),  "+v"(b[7]),
                      "+v"(b[8]), "+v"(b[9]), "+v"(b[10]), "+v"(b[11]));
#pragma unroll
    for (int nt = 0; nt < 4; ++nt) {
      accQ[nt] = WMMA_BF16(a, b[nt],     accQ[nt]);
      accK[nt] = WMMA_BF16(a, b[4 + nt], accK[nt]);
      accV[nt] = WMMA_BF16(a, b[8 + nt], accV[nt]);
    }
  }
  // C/D layout: vgpr g, lane -> row = 16*wv + g + 8*half, col = nt*16 + l15
#pragma unroll
  for (int nt = 0; nt < 4; ++nt) {
#pragma unroll
    for (int g = 0; g < 8; ++g) {
      int c = wv * 16 + g + half * 8;
      int h = nt * 16 + l15;
      Qb[c * HDIM + h] = (bf16)accQ[nt][g];
      Kb[c * HDIM + h] = (bf16)accK[nt][g];
      VT[h * CP + c]   = (bf16)accV[nt][g];   // store V transposed for P@V
    }
  }
  __syncthreads();

  // ---- Phase 2: S = Q @ K^T, scaled + masked softmax ----------------------
  v8f s[8];
#pragma unroll
  for (int t = 0; t < 8; ++t) s[t] = vz;
  v16bf aq0 = load_a16(Qb, HDIM, wv * 16, 0,  lane);
  v16bf aq1 = load_a16(Qb, HDIM, wv * 16, 32, lane);
  asm volatile("" : "+v"(aq0), "+v"(aq1));
#pragma unroll
  for (int ct = 0; ct < 8; ct += 2) {
    v16bf b00 = load_b16(Kb, HDIM, ct * 16,       0,  lane);
    v16bf b01 = load_b16(Kb, HDIM, ct * 16,       32, lane);
    v16bf b10 = load_b16(Kb, HDIM, (ct + 1) * 16, 0,  lane);
    v16bf b11 = load_b16(Kb, HDIM, (ct + 1) * 16, 32, lane);
    asm volatile("" : "+v"(b00), "+v"(b01), "+v"(b10), "+v"(b11));
    s[ct]     = WMMA_BF16(aq0, b00, s[ct]);
    s[ct]     = WMMA_BF16(aq1, b01, s[ct]);
    s[ct + 1] = WMMA_BF16(aq0, b10, s[ct + 1]);
    s[ct + 1] = WMMA_BF16(aq1, b11, s[ct + 1]);
  }
#pragma unroll
  for (int ct = 0; ct < 8; ++ct) {
    int col = ct * 16 + l15;
#pragma unroll
    for (int g = 0; g < 8; ++g) {
      float v = s[ct][g] * 0.125f;          // 1/sqrt(64)
      s[ct][g] = (col < CDIM) ? v : -1e30f; // mask padded key columns
    }
  }
  float inv[8];
#pragma unroll
  for (int g = 0; g < 8; ++g) {
    float mx = s[0][g];
#pragma unroll
    for (int ct = 1; ct < 8; ++ct) mx = fmaxf(mx, s[ct][g]);
#pragma unroll
    for (int off = 1; off < 16; off <<= 1) mx = fmaxf(mx, __shfl_xor(mx, off, 16));
    float sum = 0.f;
#pragma unroll
    for (int ct = 0; ct < 8; ++ct) { float e = __expf(s[ct][g] - mx); s[ct][g] = e; sum += e; }
#pragma unroll
    for (int off = 1; off < 16; off <<= 1) sum += __shfl_xor(sum, off, 16);
    inv[g] = 1.0f / sum;                    // defer normalization to output
  }
#pragma unroll
  for (int ct = 0; ct < 8; ++ct) {
#pragma unroll
    for (int g = 0; g < 8; ++g) {
      int c = wv * 16 + g + half * 8;
      Pb[c * CP + ct * 16 + l15] = (bf16)s[ct][g];
    }
  }
  __syncthreads();

  // ---- Phase 3: O = P @ V (4 k-steps over c', 4 h-tiles) ------------------
  v8f o[4];
#pragma unroll
  for (int t = 0; t < 4; ++t) o[t] = vz;
#pragma unroll
  for (int ks = 0; ks < 4; ++ks) {
    v16bf a = load_a16(Pb, CP, wv * 16, ks * 32, lane);
    v16bf bv[4];
#pragma unroll
    for (int ht = 0; ht < 4; ++ht)
      bv[ht] = load_b16(VT, CP, ht * 16, ks * 32, lane);
    asm volatile("" : "+v"(a), "+v"(bv[0]), "+v"(bv[1]), "+v"(bv[2]), "+v"(bv[3]));
#pragma unroll
    for (int ht = 0; ht < 4; ++ht)
      o[ht] = WMMA_BF16(a, bv[ht], o[ht]);
  }
  // normalize (same row layout as S) and stage fp32 result in LDS (alias Xb)
#pragma unroll
  for (int ht = 0; ht < 4; ++ht) {
#pragma unroll
    for (int g = 0; g < 8; ++g) {
      int c = wv * 16 + g + half * 8;
      if (c < CDIM) Ob[c * HDIM + ht * 16 + l15] = o[ht][g] * inv[g];
    }
  }
  __syncthreads();

  // ---- Drain O via async LDS->global (gfx1250 ASYNCcnt path) --------------
  // Ob at LDS offset 0; 120*64*4 = 30720 B = 1920 x 16B chunks.
  unsigned long long gbase =
      (unsigned long long)(uintptr_t)(out + (size_t)n * (CDIM * HDIM));
  for (int i = tid; i < (CDIM * HDIM * 4) / 16; i += 256) {
    unsigned long long ga = gbase + (unsigned long long)i * 16ull;
    unsigned loff = (unsigned)(i * 16);
    asm volatile("global_store_async_from_lds_b128 %0, %1, off"
                 :: "v"(ga), "v"(loff) : "memory");
  }
  asm volatile("s_wait_asynccnt 0" ::: "memory");
}

// ---------------------------------------------------------------------------
extern "C" void kernel_launch(void* const* d_in, const int* in_sizes, int n_in,
                              void* d_out, int out_size, void* d_ws, size_t ws_size,
                              hipStream_t stream) {
  const float* x  = (const float*)d_in[0];
  const float* Wk = (const float*)d_in[1];
  const float* Wq = (const float*)d_in[2];
  const float* Wv = (const float*)d_in[3];
  float* out = (float*)d_out;
  bf16* wt = (bf16*)d_ws;   // 3 * 64 * 480 bf16 = 184320 bytes

  int wtElems = 3 * HDIM * LP;
  wt_transpose_kernel<<<(wtElems + 255) / 256, 256, 0, stream>>>(Wk, Wq, Wv, wt);

  size_t smemBytes = (size_t)(CP * LP + 3 * CP * HDIM + CP * CP) * sizeof(bf16); // 204800
  attn_kernel<<<NGRP, 256, smemBytes, stream>>>(x, wt, out);
}
